// GATDecoder_39419209842694
// MI455X (gfx1250) — matrix operand order repaired and checked by reading, not compiled, not run on hardware
//
#include <hip/hip_runtime.h>

typedef __attribute__((ext_vector_type(16))) _Float16 v16h;
typedef __attribute__((ext_vector_type(8)))  float    v8f;
typedef unsigned int u32x4 __attribute__((ext_vector_type(4)));
typedef int          i32x8 __attribute__((ext_vector_type(8)));
typedef int          i32x4 __attribute__((ext_vector_type(4)));

union HF16 { v16h v; _Float16 e[16]; };
union F8   { v8f  v; float    e[8];  };

// ---------------------------------------------------------------------------
// TDM tile fetch: DMA a [tile_rows x 32]-element f16 tile (row length 32,
// row-to-row stride = row_len elements) from global into LDS at lds_off.
// Pad feature inserts 4 DWORDs after every 16 DWORDs stored -> 40-half LDS
// row stride (80 B = 5x16 B, keeps ds_load_b128 alignment).
// rows_rem = remaining rows in the tensor from this tile's start; TDM
// zero-fills reads beyond it (handles the M-tail for free).
// ---------------------------------------------------------------------------
__device__ __forceinline__ void tdm_load_tile_f16(const _Float16* gsrc,
                                                  unsigned int lds_off,
                                                  int rows_rem, int row_len,
                                                  int tile_rows) {
    const unsigned long long ga = (unsigned long long)(uintptr_t)gsrc;
    u32x4 g0;
    g0[0] = 1u;                                   // count=1, user mode
    g0[1] = lds_off;                              // lds_addr (bytes)
    g0[2] = (unsigned int)ga;                     // global_addr[31:0]
    g0[3] = (unsigned int)((ga >> 32) & 0x1FFFFFFu) | (2u << 30); // [56:32]|type=2
    i32x8 g1;
    g1[0] = (int)((1u << 16)      // data_size = 1 (2-byte elements)
                | (1u << 20)      // pad_enable
                | (3u << 22)      // pad_interval: 16 DWORDs (one 64B row)
                | (3u << 25));    // pad_amount  : 4 DWORDs (16 B)
    g1[1] = (int)(((unsigned)row_len & 0xFFFFu) << 16);           // tdim0[15:0]
    g1[2] = (int)((((unsigned)row_len >> 16) & 0xFFFFu)
                | (((unsigned)rows_rem & 0xFFFFu) << 16));        // tdim0 hi | tdim1 lo
    g1[3] = (int)((((unsigned)rows_rem >> 16) & 0xFFFFu)
                | (32u << 16));                                   // tdim1 hi | tile_dim0=32
    g1[4] = (int)((unsigned)tile_rows & 0xFFFFu);                 // tile_dim1
    g1[5] = (int)(unsigned)row_len;                               // tdim0_stride[31:0]
    g1[6] = 0;
    g1[7] = 0;
    i32x4 z4 = {0, 0, 0, 0};
#if defined(__clang_major__) && (__clang_major__ >= 23)
    i32x8 z8 = {0, 0, 0, 0, 0, 0, 0, 0};
    __builtin_amdgcn_tensor_load_to_lds(g0, g1, z4, z4, z8, 0);
#else
    __builtin_amdgcn_tensor_load_to_lds(g0, g1, z4, z4, 0);
#endif
}

// ---------------------------------------------------------------------------
// WMMA GEMM, f16 operands / f32 accumulate: C[M,N] = A[M,K] x Bt^T (+bias).
// A  : f16 [M,K] row-major.   Bt : f16 [N,K] row-major (pre-transposed).
// Block = 256 threads = 8 waves (4 M-slices x 2 N-slices); block tile 64x32.
// Tiles staged by the Tensor Data Mover, double-buffered in LDS; wave 0
// issues tile s+1 while all waves run v_wmma on tile s.
// Requires N % 32 == 0, K % 32 == 0.
// ---------------------------------------------------------------------------
__global__ void wmma_gemm_tdm(const _Float16* __restrict__ A,
                              const _Float16* __restrict__ Bt,
                              const float* __restrict__ bias,
                              float* __restrict__ C,
                              int M, int N, int K) {
    __shared__ _Float16 As[2][64][40];   // padded: 80 B row stride
    __shared__ _Float16 Bs[2][32][40];

    const int tid  = threadIdx.x;
    const int wave = tid >> 5;
    const int lane = tid & 31;
    const int wm   = wave >> 1;
    const int wn   = wave & 1;
    const int m0   = blockIdx.x * 64;
    const int n0   = blockIdx.y * 32;
    const int steps = K >> 5;

    const int rows_remA = M - m0;        // TDM zero-fills rows beyond this
    const int rows_remB = N - n0;

    v8f acc = {};

    if (wave == 0) {
        tdm_load_tile_f16(A  + (size_t)m0 * K, (unsigned)(uintptr_t)&As[0][0][0],
                          rows_remA, K, 64);
        tdm_load_tile_f16(Bt + (size_t)n0 * K, (unsigned)(uintptr_t)&Bs[0][0][0],
                          rows_remB, K, 32);
    }

    for (int s = 0; s < steps; ++s) {
        const int p = s & 1;
        if (wave == 0) {
            if (s + 1 < steps) {
                const int k1 = (s + 1) << 5;
                tdm_load_tile_f16(A  + (size_t)m0 * K + k1,
                                  (unsigned)(uintptr_t)&As[1 - p][0][0],
                                  rows_remA, K, 64);
                tdm_load_tile_f16(Bt + (size_t)n0 * K + k1,
                                  (unsigned)(uintptr_t)&Bs[1 - p][0][0],
                                  rows_remB, K, 32);
                __builtin_amdgcn_s_wait_tensorcnt((short)2); // tile s landed
            } else {
                __builtin_amdgcn_s_wait_tensorcnt((short)0);
            }
        }
        __syncthreads();                 // publish tile s to all waves

        // A 16x32 fragment: lane L -> row M=L&15; half j: K=16*(L>>4)+j.
        HF16 af, bf;
        const int am   = wm * 16 + (lane & 15);
        const int akhi = (lane >> 4) * 16;
        #pragma unroll
        for (int i = 0; i < 8; ++i) {
            af.e[i]     = As[p][am][akhi + i];
            af.e[8 + i] = As[p][am][akhi + 8 + i];
        }
        // B 32x16 fragment: lane L -> col N=L&15; half j: K=16*(L>>4)+j.
        const int bn   = wn * 16 + (lane & 15);
        const int bkhi = (lane >> 4) * 16;
        #pragma unroll
        for (int i = 0; i < 16; ++i)
            bf.e[i] = Bs[p][bn][bkhi + i];

        acc = __builtin_amdgcn_wmma_f32_16x16x32_f16(
                  false, af.v, false, bf.v, (short)0, acc, false, false);
        __syncthreads();                 // reads of buffer p done before reuse
    }

    // C/D layout: VGPR r -> M = r + 8*(lane>=16), N = lane&15.
    F8 res; res.v = acc;
    const int cn = n0 + wn * 16 + (lane & 15);
    const float bv = bias ? bias[cn] : 0.0f;
    #pragma unroll
    for (int r = 0; r < 8; ++r) {
        const int cm = m0 + wm * 16 + r + ((lane >> 4) * 8);
        if (cm < M) C[(size_t)cm * N + cn] = res.e[r] + bv;
    }
}

// ---------------------------------------------------------------------------
// f32 -> f16 conversions (optionally NaN -> -100, per reference fixup)
// ---------------------------------------------------------------------------
__global__ void cvt_f16(_Float16* __restrict__ dst, const float* __restrict__ src,
                        int n, int nanfix) {
    const int i = blockIdx.x * blockDim.x + threadIdx.x;
    if (i >= n) return;
    float v = src[i];
    if (nanfix && v != v) v = -100.0f;
    dst[i] = (_Float16)v;
}

// src[R][Ccols] (f32) -> dst[Ccols][R] (f16)
__global__ void cvt_f16_t(_Float16* __restrict__ dst, const float* __restrict__ src,
                          int R, int Ccols) {
    const int i = blockIdx.x * blockDim.x + threadIdx.x;
    if (i >= R * Ccols) return;
    const int r = i / Ccols, c = i - r * Ccols;
    dst[c * R + r] = (_Float16)src[i];
}

// ---------------------------------------------------------------------------
// Attention scores: one wave32 per (node, head); shuffle tree reduction.
// ---------------------------------------------------------------------------
__global__ void att_scores(const float* __restrict__ h,
                           const float* __restrict__ att_src,
                           const float* __restrict__ att_dst,
                           float* __restrict__ a_s, float* __restrict__ a_d,
                           int N, int H, int C) {
    const int gw   = (blockIdx.x * blockDim.x + threadIdx.x) >> 5;
    const int lane = threadIdx.x & 31;
    if (gw >= N * H) return;
    const int n  = gw / H;
    const int hh = gw - n * H;
    const float* hp = h + ((size_t)n * H + hh) * C;
    const float* sp = att_src + hh * C;
    const float* dp = att_dst + hh * C;
    float ss = 0.f, sd = 0.f;
    for (int c = lane; c < C; c += 32) {
        const float v = hp[c];
        ss += v * sp[c];
        sd += v * dp[c];
    }
    #pragma unroll
    for (int o = 16; o > 0; o >>= 1) {
        ss += __shfl_xor(ss, o, 32);
        sd += __shfl_xor(sd, o, 32);
    }
    if (lane == 0) { a_s[gw] = ss; a_d[gw] = sd; }
}

__global__ void fill_f32(float* __restrict__ p, float v, int n) {
    const int i = blockIdx.x * blockDim.x + threadIdx.x;
    if (i < n) p[i] = v;
}

__global__ void init_bias(float* __restrict__ out, const float* __restrict__ bias,
                          int total, int F) {
    const int i = blockIdx.x * blockDim.x + threadIdx.x;
    if (i < total) out[i] = bias[i % F];
}

__device__ __forceinline__ float leaky02(float v) { return v > 0.f ? v : 0.2f * v; }

__device__ __forceinline__ void atomic_max_f32(float* addr, float val) {
    if (val >= 0.f) atomicMax((int*)addr, __float_as_int(val));
    else            atomicMin((unsigned int*)addr, __float_as_uint(val));
}

__global__ void edge_max(const int* __restrict__ src, const int* __restrict__ dst,
                         const float* __restrict__ a_s, const float* __restrict__ a_d,
                         float* __restrict__ mmax, int E, int H) {
    const int i = blockIdx.x * blockDim.x + threadIdx.x;
    if (i >= E * H) return;
    const int e = i / H, hh = i - e * H;
    const int s = src[e], d = dst[e];
    atomic_max_f32(&mmax[d * H + hh], leaky02(a_s[s * H + hh] + a_d[d * H + hh]));
}

__global__ void edge_denom(const int* __restrict__ src, const int* __restrict__ dst,
                           const float* __restrict__ a_s, const float* __restrict__ a_d,
                           const float* __restrict__ mmax, float* __restrict__ den,
                           int E, int H) {
    const int i = blockIdx.x * blockDim.x + threadIdx.x;
    if (i >= E * H) return;
    const int e = i / H, hh = i - e * H;
    const int s = src[e], d = dst[e];
    const float v = leaky02(a_s[s * H + hh] + a_d[d * H + hh]);
    atomicAdd(&den[d * H + hh], __expf(v - mmax[d * H + hh]));
}

__global__ void edge_agg(const int* __restrict__ src, const int* __restrict__ dst,
                         const float* __restrict__ h,
                         const float* __restrict__ a_s, const float* __restrict__ a_d,
                         const float* __restrict__ mmax, const float* __restrict__ den,
                         float* __restrict__ out, int H, int C) {
    const int e  = blockIdx.x;
    const int t  = threadIdx.x;
    const int hh = t / C;
    const int s = src[e], d = dst[e];
    const float v = leaky02(a_s[s * H + hh] + a_d[d * H + hh]);
    const float alpha = __expf(v - mmax[d * H + hh]) / den[d * H + hh];
    const int F = H * C;
    atomicAdd(&out[(size_t)d * F + t], h[(size_t)s * F + t] * alpha);
}

// ---------------------------------------------------------------------------
extern "C" void kernel_launch(void* const* d_in, const int* in_sizes, int n_in,
                              void* d_out, int out_size, void* d_ws, size_t ws_size,
                              hipStream_t stream) {
    const int IN = 128;
    const int H1 = 4, C1 = 64,  F1 = H1 * C1;    // layer 1
    const int H2 = 1, C2 = 128, F2 = H2 * C2;    // layer 2
    const int N  = in_sizes[0] / IN;             // 50000
    const int E  = in_sizes[1] / 2;              // 800000

    const float* x        = (const float*)d_in[0];
    const int*   src      = (const int*)d_in[1];
    const int*   dst      = src + E;
    const float* W1       = (const float*)d_in[3];
    const float* att_src1 = (const float*)d_in[4];
    const float* att_dst1 = (const float*)d_in[5];
    const float* b1       = (const float*)d_in[6];
    const float* W2       = (const float*)d_in[7];
    const float* att_src2 = (const float*)d_in[8];
    const float* att_dst2 = (const float*)d_in[9];
    const float* b2       = (const float*)d_in[10];
    const float* head_w   = (const float*)d_in[11];
    const float* head_b   = (const float*)d_in[12];
    float*       out      = (float*)d_out;

    // ---- workspace carving ----
    float* ws   = (float*)d_ws;
    float* h1   = ws;                          // [N,256] f32
    float* agg1 = h1 + (size_t)N * F1;         // [N,256] f32
    float* a_s1 = agg1 + (size_t)N * F1;
    float* a_d1 = a_s1 + (size_t)N * H1;
    float* m1   = a_d1 + (size_t)N * H1;
    float* den1 = m1   + (size_t)N * H1;
    float* a_s2 = den1 + (size_t)N * H1;
    float* a_d2 = a_s2 + (size_t)N * H2;
    float* m2   = a_d2 + (size_t)N * H2;
    float* den2 = m2   + (size_t)N * H2;
    float* h2   = h1;                          // layer-2 reuse of h1 region
    float* agg2 = h1 + (size_t)N * F2;
    // f16 region
    _Float16* fh  = (_Float16*)(den2 + (size_t)N * H2);
    _Float16* xh  = fh;                        // [N,128]
    _Float16* a1h = xh  + (size_t)N * IN;      // [N,256]
    _Float16* a2h = a1h + (size_t)N * F1;      // [N,128]
    _Float16* w1t = a2h + (size_t)N * F2;      // [256,128]
    _Float16* w2t = w1t + (size_t)IN * F1;     // [128,256]
    _Float16* hwh = w2t + (size_t)F1 * F2;     // [128,128]

    const float NEG_INF = -__builtin_inff();
    const int MT = (N + 63) / 64;

    // ---- layer 1 ----
    cvt_f16<<<(N * IN + 255) / 256, 256, 0, stream>>>(xh, x, N * IN, 0);
    cvt_f16_t<<<(IN * F1 + 255) / 256, 256, 0, stream>>>(w1t, W1, IN, F1);
    wmma_gemm_tdm<<<dim3(MT, F1 / 32), 256, 0, stream>>>(xh, w1t, nullptr, h1, N, F1, IN);
    att_scores<<<(N * H1 * 32 + 255) / 256, 256, 0, stream>>>(h1, att_src1, att_dst1,
                                                              a_s1, a_d1, N, H1, C1);
    fill_f32<<<(N * H1 + 255) / 256, 256, 0, stream>>>(m1, NEG_INF, N * H1);
    fill_f32<<<(N * H1 + 255) / 256, 256, 0, stream>>>(den1, 0.f, N * H1);
    init_bias<<<(N * F1 + 255) / 256, 256, 0, stream>>>(agg1, b1, N * F1, F1);
    edge_max<<<(E * H1 + 255) / 256, 256, 0, stream>>>(src, dst, a_s1, a_d1, m1, E, H1);
    edge_denom<<<(E * H1 + 255) / 256, 256, 0, stream>>>(src, dst, a_s1, a_d1, m1, den1, E, H1);
    edge_agg<<<E, F1, 0, stream>>>(src, dst, h1, a_s1, a_d1, m1, den1, agg1, H1, C1);

    // ---- layer 2 ----
    cvt_f16<<<(N * F1 + 255) / 256, 256, 0, stream>>>(a1h, agg1, N * F1, 0);
    cvt_f16_t<<<(F1 * F2 + 255) / 256, 256, 0, stream>>>(w2t, W2, F1, F2);
    wmma_gemm_tdm<<<dim3(MT, F2 / 32), 256, 0, stream>>>(a1h, w2t, nullptr, h2, N, F2, F1);
    att_scores<<<(N * H2 * 32 + 255) / 256, 256, 0, stream>>>(h2, att_src2, att_dst2,
                                                              a_s2, a_d2, N, H2, C2);
    fill_f32<<<(N * H2 + 255) / 256, 256, 0, stream>>>(m2, NEG_INF, N * H2);
    fill_f32<<<(N * H2 + 255) / 256, 256, 0, stream>>>(den2, 0.f, N * H2);
    init_bias<<<(N * F2 + 255) / 256, 256, 0, stream>>>(agg2, b2, N * F2, F2);
    edge_max<<<(E * H2 + 255) / 256, 256, 0, stream>>>(src, dst, a_s2, a_d2, m2, E, H2);
    edge_denom<<<(E * H2 + 255) / 256, 256, 0, stream>>>(src, dst, a_s2, a_d2, m2, den2, E, H2);
    edge_agg<<<E, F2, 0, stream>>>(src, dst, h2, a_s2, a_d2, m2, den2, agg2, H2, C2);

    // ---- head: out = nanfix(agg2) @ head_w^T + head_b ----
    cvt_f16<<<(N * F2 + 255) / 256, 256, 0, stream>>>(a2h, agg2, N * F2, 1);
    cvt_f16<<<(F2 * F2 + 255) / 256, 256, 0, stream>>>(hwh, head_w, F2 * F2, 0);
    wmma_gemm_tdm<<<dim3(MT, F2 / 32), 256, 0, stream>>>(a2h, hwh, head_b, out, N, F2, F2);
}
